// MPNN_45543833207154
// MI455X (gfx1250) — compile-verified, matrix operand order checked by reading, not run on hardware
//
#include <hip/hip_runtime.h>
#include <hip/hip_bf16.h>

// ---------------- problem constants (match reference) ----------------
#define BATCH   64
#define NNODE   32
#define NFEAT   16
#define HID     64
#define MSGD    64
#define TGT     12
#define ROWS    (BATCH * NNODE)          // 2048 node rows
#define EDGES   (BATCH * NNODE * NNODE)  // 65536 edge rows
#define KC      4160                     // 4096 (w,k) + 32 (u/g fold) + 32 zero pad

typedef _Float16 h16v __attribute__((ext_vector_type(16)));
typedef _Float16 h8v  __attribute__((ext_vector_type(8)));
typedef float    f8v  __attribute__((ext_vector_type(8)));

union H16U { h16v v; h8v c[2]; };

// Unconditional fragment load for row-major [rows,K] f16 (K%32==0, rows padded
// so every tile row is valid). CDNA5 16-bit A/B layout: lane L holds row L%16;
// halves 0-7 = K k0+(L/16)*8.., halves 8-15 = K k0+16+(L/16)*8..
__device__ __forceinline__ h16v MPNN_frag_load(const _Float16* __restrict__ p, long off) {
  H16U f;
  f.c[0] = *(const h8v*)(p + off);
  f.c[1] = *(const h8v*)(p + off + 16);
  return f.v;
}

#define ACT_NONE 0
#define ACT_RELU 1
#define ST_F16   0   // f16 out[M,N] row-major
#define ST_F32   1   // f32 out[M,N] row-major
#define ST_TMODE 2   // T[b,w,k,m] -> Tt[b][m][w*128+k]      (f16)
#define ST_UMODE 3   // u[b,w,m]   -> Tt[b][m][4096+w]       (f16)

// Tiled WMMA GEMM: D = act(A[M,K] x Bt[NB,K]^T + bias). One wave owns a 16x32
// output tile (1 A frag, 2 B frags, 2 accumulators), K-loop step 32 with
// next-iteration fragment prefetch. NB = padded #B rows (mult of 32), N =
// logical column count for bias/stores. blockIdx.y batches via strides.
__global__ void MPNN_wmma_gemm(const _Float16* __restrict__ A,
                               const _Float16* __restrict__ Bt,
                               const float* __restrict__ bias,
                               void* __restrict__ out,
                               int M, int N, int NB, int K,
                               long sA, long sB, long sO,
                               int act, int mode) {
  const int wave = threadIdx.x >> 5;
  const int lane = threadIdx.x & 31;
  const int sub  = lane >> 4;
  const int lidx = lane & 15;
  const int tilesN2 = NB >> 5;                  // 16x32 tiles along N
  const int tilesM  = M >> 4;                   // M always a multiple of 16
  const int tile = blockIdx.x * (blockDim.x >> 5) + wave;
  if (tile >= tilesM * tilesN2) return;         // wave-uniform exit
  const int tm  = tile / tilesN2;
  const int tn2 = tile - tm * tilesN2;
  const long batch = blockIdx.y;
  A  += batch * sA;
  Bt += batch * sB;

  const long aoff  = (long)(tm * 16 + lidx) * K + sub * 8;
  const long b0off = (long)(tn2 * 32 + lidx) * K + sub * 8;
  const long b1off = b0off + (long)16 * K;

  f8v acc0 = {}, acc1 = {};
  h16v a  = MPNN_frag_load(A,  aoff);
  h16v b0 = MPNN_frag_load(Bt, b0off);
  h16v b1 = MPNN_frag_load(Bt, b1off);
  for (int k0 = 32; k0 < K; k0 += 32) {
    h16v an  = MPNN_frag_load(A,  aoff  + k0);   // prefetch next step
    h16v b0n = MPNN_frag_load(Bt, b0off + k0);
    h16v b1n = MPNN_frag_load(Bt, b1off + k0);
    acc0 = __builtin_amdgcn_wmma_f32_16x16x32_f16(false, a, false, b0,
                                                  (short)0, acc0, false, false);
    acc1 = __builtin_amdgcn_wmma_f32_16x16x32_f16(false, a, false, b1,
                                                  (short)0, acc1, false, false);
    a = an; b0 = b0n; b1 = b1n;
  }
  acc0 = __builtin_amdgcn_wmma_f32_16x16x32_f16(false, a, false, b0,
                                                (short)0, acc0, false, false);
  acc1 = __builtin_amdgcn_wmma_f32_16x16x32_f16(false, a, false, b1,
                                                (short)0, acc1, false, false);

  f8v accs[2] = {acc0, acc1};
#pragma unroll
  for (int t = 0; t < 2; ++t) {
    const int col = tn2 * 32 + t * 16 + lidx;
    if (col >= N) continue;                     // after all WMMA -> safe
    const float bv = bias ? bias[col] : 0.0f;
#pragma unroll
    for (int i = 0; i < 8; ++i) {
      const int row = tm * 16 + i + sub * 8;
      float v = accs[t][i] + bv;
      if (act == ACT_RELU) v = fmaxf(v, 0.0f);
      if (mode == ST_F16) {
        ((_Float16*)out)[batch * sO + (long)row * N + col] = (_Float16)v;
      } else if (mode == ST_F32) {
        ((float*)out)[batch * sO + (long)row * N + col] = v;
      } else if (mode == ST_TMODE) {
        // row = b*32+w ; col = k*64+m  ->  Tt[b][m][w*128+k]
        const int bb = row >> 5, w = row & 31, kk = col >> 6, mm = col & 63;
        ((_Float16*)out)[((long)bb * 64 + mm) * KC + w * 128 + kk] = (_Float16)v;
      } else {
        // row = b*32+w ; col = m (N==64)  ->  Tt[b][m][4096+w]
        const int bb = row >> 5, w = row & 31;
        ((_Float16*)out)[((long)bb * 64 + col) * KC + 4096 + w] = (_Float16)v;
      }
    }
  }
}

// ---------------- small helper kernels ----------------
__global__ void MPNN_f32_to_f16(const float* __restrict__ s, _Float16* __restrict__ d, long n) {
  long i = (long)blockIdx.x * blockDim.x + threadIdx.x;
  if (i < n) d[i] = (_Float16)s[i];
}

// e[EDGES,8] f32 -> e16p[EDGES,64] f16 zero-padded (K=8 -> K=64).
__global__ void MPNN_pad_e(const float* __restrict__ e, _Float16* __restrict__ d) {
  long i = (long)blockIdx.x * blockDim.x + threadIdx.x;
  if (i >= (long)EDGES * 64) return;
  long r = i >> 6; int j = (int)(i & 63);
  d[i] = (j < 8) ? (_Float16)e[r * 8 + j] : (_Float16)0.0f;
}

// W[K,N] f32 (x@W convention) -> Wt[NB,KB] f16, transposed + zero padded.
__global__ void MPNN_prep_B(const float* __restrict__ W, _Float16* __restrict__ Wt,
                            int K, int N, int KB, int NB) {
  long i = (long)blockIdx.x * blockDim.x + threadIdx.x;
  if (i >= (long)NB * KB) return;
  int n = (int)(i / KB), k = (int)(i - (long)n * KB);
  Wt[i] = (n < N && k < K) ? (_Float16)W[(long)k * N + n] : (_Float16)0.0f;
}

__global__ void MPNN_init(const float* __restrict__ h_in, float* __restrict__ h0,
                          float* __restrict__ h, _Float16* __restrict__ h016,
                          _Float16* __restrict__ h16, float* __restrict__ mask) {
  int i = blockIdx.x * blockDim.x + threadIdx.x;
  if (i >= ROWS * HID) return;
  int r = i >> 6, j = i & 63;
  float v = (j < NFEAT) ? h_in[r * NFEAT + j] : 0.0f;
  h0[i] = v; h[i] = v;
  _Float16 vh = (_Float16)v;
  h016[i] = vh; h16[i] = vh;
  if (j == 0) {
    float s = 0.0f;
    for (int t = 0; t < NFEAT; ++t) s += h_in[r * NFEAT + t];
    mask[r] = (s > 0.0f) ? 1.0f : 0.0f;
  }
}

// S[b][v][w*128+k] = g*relu3 ; S[b][v][4096+w] = g ; S[b][v][4128..4159] = 0
__global__ void MPNN_build_S(const float* __restrict__ g, const _Float16* __restrict__ relu3,
                             _Float16* __restrict__ S) {
  long i = (long)blockIdx.x * blockDim.x + threadIdx.x;
  if (i >= (long)BATCH * NNODE * KC) return;
  int col = (int)(i % KC);
  long rv = i / KC;                  // b*32+v
  if (col < 4096) {
    int w = col >> 7, k = col & 127;
    float gg = g[rv * NNODE + w];
    S[i] = (_Float16)(gg * (float)relu3[(rv * NNODE + w) * 128 + k]);
  } else if (col < 4128) {
    S[i] = (_Float16)g[rv * NNODE + (col - 4096)];
  } else {
    S[i] = (_Float16)0.0f;           // pad -> makes KC a multiple of 64
  }
}

__global__ void MPNN_gru(const float* __restrict__ gi, const float* __restrict__ gh,
                         const float* __restrict__ mask,
                         float* __restrict__ h, _Float16* __restrict__ h16) {
  int i = blockIdx.x * blockDim.x + threadIdx.x;
  if (i >= ROWS * HID) return;
  int r = i >> 6, j = i & 63;
  float ir = gi[r * 192 + j],        hr = gh[r * 192 + j];
  float iz = gi[r * 192 + 64 + j],   hz = gh[r * 192 + 64 + j];
  float in_ = gi[r * 192 + 128 + j], hn = gh[r * 192 + 128 + j];
  float rg = 1.0f / (1.0f + __expf(-(ir + hr)));
  float z  = 1.0f / (1.0f + __expf(-(iz + hz)));
  float n  = tanhf(in_ + rg * hn);
  float hv = ((1.0f - z) * n + z * h[i]) * mask[r];
  h[i] = hv;
  h16[i] = (_Float16)hv;
}

__global__ void MPNN_cat(const _Float16* __restrict__ h16, const _Float16* __restrict__ h016,
                         _Float16* __restrict__ cat) {
  int i = blockIdx.x * blockDim.x + threadIdx.x;
  if (i >= ROWS * 128) return;
  int r = i >> 7, j = i & 127;
  cat[i] = (j < 64) ? h16[r * 64 + j] : h016[r * 64 + (j - 64)];
}

__global__ void MPNN_readout(const float* __restrict__ gate, const float* __restrict__ val,
                             const float* __restrict__ mask, float* __restrict__ out) {
  int i = blockIdx.x * blockDim.x + threadIdx.x;
  if (i >= BATCH * TGT) return;
  int b = i / TGT, t = i - b * TGT;
  float s = 0.0f;
  for (int v = 0; v < NNODE; ++v) {
    int r = b * NNODE + v;
    float gsig = 1.0f / (1.0f + __expf(-gate[r * TGT + t]));
    s += gsig * val[r * TGT + t] * mask[r];
  }
  out[i] = s;
}

// ---------------- host side ----------------
static inline void launch_gemm(hipStream_t st, const _Float16* A, const _Float16* Bt,
                               const float* bias, void* out, int M, int N, int NB, int K,
                               int batches, long sA, long sB, long sO, int act, int mode) {
  int tiles = (M / 16) * (NB / 32);
  dim3 grid((tiles + 3) / 4, batches, 1);
  MPNN_wmma_gemm<<<grid, 128, 0, st>>>(A, Bt, bias, out, M, N, NB, K, sA, sB, sO, act, mode);
}
static inline dim3 ewgrid(long n) { return dim3((unsigned)((n + 255) / 256)); }

extern "C" void kernel_launch(void* const* d_in, const int* in_sizes, int n_in,
                              void* d_out, int out_size, void* d_ws, size_t ws_size,
                              hipStream_t stream) {
  (void)in_sizes; (void)n_in; (void)out_size; (void)ws_size;
  const float* g    = (const float*)d_in[0];
  const float* h_in = (const float*)d_in[1];
  const float* e    = (const float*)d_in[2];
  const float* eW0  = (const float*)d_in[3];  const float* eb0 = (const float*)d_in[4];
  const float* eW1  = (const float*)d_in[5];  const float* eb1 = (const float*)d_in[6];
  const float* eW2  = (const float*)d_in[7];  const float* eb2 = (const float*)d_in[8];
  const float* eW3  = (const float*)d_in[9];  const float* eb3 = (const float*)d_in[10];
  const float* Wih  = (const float*)d_in[11]; const float* Whh = (const float*)d_in[12];
  const float* bih  = (const float*)d_in[13]; const float* bhh = (const float*)d_in[14];
  const float* riW0 = (const float*)d_in[15]; const float* rib0 = (const float*)d_in[16];
  const float* riW1 = (const float*)d_in[17]; const float* rib1 = (const float*)d_in[18];
  const float* riW2 = (const float*)d_in[19]; const float* rib2 = (const float*)d_in[20];
  const float* riW3 = (const float*)d_in[21]; const float* rib3 = (const float*)d_in[22];
  const float* rjW0 = (const float*)d_in[23]; const float* rjb0 = (const float*)d_in[24];
  const float* rjW1 = (const float*)d_in[25]; const float* rjb1 = (const float*)d_in[26];
  const float* rjW2 = (const float*)d_in[27]; const float* rjb2 = (const float*)d_in[28];
  const float* rjW3 = (const float*)d_in[29]; const float* rjb3 = (const float*)d_in[30];
  float* out = (float*)d_out;

  // ---- workspace bump allocator (256B aligned) ----
  char* base = (char*)d_ws; size_t off = 0;
  auto alloc = [&](size_t bytes) -> void* {
    void* p = base + off; off += (bytes + 255) & ~(size_t)255; return p;
  };
  _Float16* e16p  = (_Float16*)alloc((size_t)EDGES * 64 * 2);   // K padded 8->64
  _Float16* y1    = (_Float16*)alloc((size_t)EDGES * 128 * 2);
  _Float16* y2    = (_Float16*)alloc((size_t)EDGES * 256 * 2);
  _Float16* relu3 = (_Float16*)alloc((size_t)EDGES * 128 * 2);
  _Float16* eW0t  = (_Float16*)alloc(128 * 64 * 2);             // [128,64] padded
  _Float16* eW1t  = (_Float16*)alloc(256 * 128 * 2);
  _Float16* eW2t  = (_Float16*)alloc(128 * 256 * 2);
  _Float16* eW3h  = (_Float16*)alloc(8192 * 64 * 2);            // reinterpret [128,4096]->[8192,64]
  _Float16* b3h   = (_Float16*)alloc(64 * 64 * 2);              // reinterpret [4096]->[64,64]
  _Float16* Wih16 = (_Float16*)alloc(192 * 64 * 2);             // already [N,K]
  _Float16* Whh16 = (_Float16*)alloc(192 * 64 * 2);
  _Float16* riW0t = (_Float16*)alloc(128 * 128 * 2);
  _Float16* riW1t = (_Float16*)alloc(256 * 128 * 2);
  _Float16* riW2t = (_Float16*)alloc(128 * 256 * 2);
  _Float16* riW3t = (_Float16*)alloc(32 * 128 * 2);             // N padded 12->32
  _Float16* rjW0t = (_Float16*)alloc(128 * 64 * 2);
  _Float16* rjW1t = (_Float16*)alloc(256 * 128 * 2);
  _Float16* rjW2t = (_Float16*)alloc(128 * 256 * 2);
  _Float16* rjW3t = (_Float16*)alloc(32 * 128 * 2);             // N padded 12->32
  _Float16* h16   = (_Float16*)alloc((size_t)ROWS * HID * 2);
  _Float16* h016  = (_Float16*)alloc((size_t)ROWS * HID * 2);
  _Float16* m16   = (_Float16*)alloc((size_t)ROWS * MSGD * 2);
  _Float16* S     = (_Float16*)alloc((size_t)BATCH * NNODE * KC * 2);
  _Float16* Tt    = (_Float16*)alloc((size_t)BATCH * 64 * KC * 2);
  _Float16* cat16 = (_Float16*)alloc((size_t)ROWS * 128 * 2);
  _Float16* r1    = (_Float16*)alloc((size_t)ROWS * 128 * 2);
  _Float16* r2    = (_Float16*)alloc((size_t)ROWS * 256 * 2);
  _Float16* r3    = (_Float16*)alloc((size_t)ROWS * 128 * 2);
  float* h0   = (float*)alloc((size_t)ROWS * HID * 4);
  float* h    = (float*)alloc((size_t)ROWS * HID * 4);
  float* mask = (float*)alloc((size_t)ROWS * 4);
  float* mmsg = (float*)alloc((size_t)ROWS * MSGD * 4);
  float* gi   = (float*)alloc((size_t)ROWS * 192 * 4);
  float* gh   = (float*)alloc((size_t)ROWS * 192 * 4);
  float* gate = (float*)alloc((size_t)ROWS * TGT * 4);
  float* val  = (float*)alloc((size_t)ROWS * TGT * 4);

  hipStream_t st = stream;
  // ---- weight / input precision prep ----
  MPNN_pad_e<<<ewgrid((long)EDGES * 64), 256, 0, st>>>(e, e16p);
  MPNN_prep_B<<<ewgrid(128 * 64), 256, 0, st>>>(eW0, eW0t, 8, 128, 64, 128);
  MPNN_prep_B<<<ewgrid(256 * 128), 256, 0, st>>>(eW1, eW1t, 128, 256, 128, 256);
  MPNN_prep_B<<<ewgrid(128 * 256), 256, 0, st>>>(eW2, eW2t, 256, 128, 256, 128);
  MPNN_f32_to_f16<<<ewgrid(524288), 256, 0, st>>>(eW3, eW3h, 524288);
  MPNN_f32_to_f16<<<ewgrid(4096), 256, 0, st>>>(eb3, b3h, 4096);
  MPNN_f32_to_f16<<<ewgrid(192 * 64), 256, 0, st>>>(Wih, Wih16, 192 * 64);
  MPNN_f32_to_f16<<<ewgrid(192 * 64), 256, 0, st>>>(Whh, Whh16, 192 * 64);
  MPNN_prep_B<<<ewgrid(128 * 128), 256, 0, st>>>(riW0, riW0t, 128, 128, 128, 128);
  MPNN_prep_B<<<ewgrid(256 * 128), 256, 0, st>>>(riW1, riW1t, 128, 256, 128, 256);
  MPNN_prep_B<<<ewgrid(128 * 256), 256, 0, st>>>(riW2, riW2t, 256, 128, 256, 128);
  MPNN_prep_B<<<ewgrid(32 * 128), 256, 0, st>>>(riW3, riW3t, 128, 12, 128, 32);
  MPNN_prep_B<<<ewgrid(128 * 64), 256, 0, st>>>(rjW0, rjW0t, 64, 128, 64, 128);
  MPNN_prep_B<<<ewgrid(256 * 128), 256, 0, st>>>(rjW1, rjW1t, 128, 256, 128, 256);
  MPNN_prep_B<<<ewgrid(128 * 256), 256, 0, st>>>(rjW2, rjW2t, 256, 128, 256, 128);
  MPNN_prep_B<<<ewgrid(32 * 128), 256, 0, st>>>(rjW3, rjW3t, 128, 12, 128, 32);
  MPNN_init<<<ewgrid(ROWS * HID), 256, 0, st>>>(h_in, h0, h, h016, h16, mask);

  // ---- edge MLP (shared across layers): 8 -> 128 -> 256 -> 128, relu ----
  launch_gemm(st, e16p, eW0t, eb0, y1,    EDGES, 128, 128, 64,  1, 0, 0, 0, ACT_RELU, ST_F16);
  launch_gemm(st, y1,   eW1t, eb1, y2,    EDGES, 256, 256, 128, 1, 0, 0, 0, ACT_RELU, ST_F16);
  launch_gemm(st, y2,   eW2t, eb2, relu3, EDGES, 128, 128, 256, 1, 0, 0, 0, ACT_RELU, ST_F16);

  // ---- 3 message-passing layers ----
  for (int layer = 0; layer < 3; ++layer) {
    // T[b,w,k,m] = h @ W3(reint [8192,64])  -> permuted into Tt[b][m][w*128+k]
    launch_gemm(st, h16, eW3h, nullptr, Tt, ROWS, 8192, 8192, 64, 1, 0, 0, 0, ACT_NONE, ST_TMODE);
    // u[b,w,m] = h @ b3(reint [64,64])     -> Tt[b][m][4096+w]
    launch_gemm(st, h16, b3h, nullptr, Tt, ROWS, 64, 64, 64, 1, 0, 0, 0, ACT_NONE, ST_UMODE);
    // S[b][v][(w,k)] = g*relu3 ; S[b][v][4096+w] = g ; pad zeros
    MPNN_build_S<<<ewgrid((long)BATCH * NNODE * KC), 256, 0, st>>>(g, relu3, S);
    // m[b,v,m] = S_b[32,KC] x Tt_b[64,KC]^T   (batched over b)
    launch_gemm(st, S, Tt, nullptr, mmsg, NNODE, MSGD, MSGD, KC, BATCH,
                (long)NNODE * KC, (long)64 * KC, (long)NNODE * MSGD, ACT_NONE, ST_F32);
    // GRU gate projections
    MPNN_f32_to_f16<<<ewgrid(ROWS * MSGD), 256, 0, st>>>(mmsg, m16, (long)ROWS * MSGD);
    launch_gemm(st, m16, Wih16, bih, gi, ROWS, 192, 192, 64, 1, 0, 0, 0, ACT_NONE, ST_F32);
    launch_gemm(st, h16, Whh16, bhh, gh, ROWS, 192, 192, 64, 1, 0, 0, 0, ACT_NONE, ST_F32);
    MPNN_gru<<<ewgrid(ROWS * HID), 256, 0, st>>>(gi, gh, mask, h, h16);
  }

  // ---- readout ----
  MPNN_cat<<<ewgrid(ROWS * 128), 256, 0, st>>>(h16, h016, cat16);
  launch_gemm(st, cat16, riW0t, rib0, r1,   ROWS, 128, 128, 128, 1, 0, 0, 0, ACT_RELU, ST_F16);
  launch_gemm(st, r1,    riW1t, rib1, r2,   ROWS, 256, 256, 128, 1, 0, 0, 0, ACT_RELU, ST_F16);
  launch_gemm(st, r2,    riW2t, rib2, r3,   ROWS, 128, 128, 256, 1, 0, 0, 0, ACT_RELU, ST_F16);
  launch_gemm(st, r3,    riW3t, rib3, gate, ROWS, 12,  32,  128, 1, 0, 0, 0, ACT_NONE, ST_F32);
  launch_gemm(st, h16,   rjW0t, rjb0, r1,   ROWS, 128, 128, 64,  1, 0, 0, 0, ACT_RELU, ST_F16);
  launch_gemm(st, r1,    rjW1t, rjb1, r2,   ROWS, 256, 256, 128, 1, 0, 0, 0, ACT_RELU, ST_F16);
  launch_gemm(st, r2,    rjW2t, rjb2, r3,   ROWS, 128, 128, 256, 1, 0, 0, 0, ACT_RELU, ST_F16);
  launch_gemm(st, r3,    rjW3t, rjb3, val,  ROWS, 12,  32,  128, 1, 0, 0, 0, ACT_NONE, ST_F32);
  MPNN_readout<<<ewgrid(BATCH * TGT), 256, 0, st>>>(gate, val, mask, out);
}
